// MultiheadAttention_59571196395811
// MI455X (gfx1250) — compile-verified, matrix operand order checked by reading, not run on hardware
//
#include <hip/hip_runtime.h>
#include <hip/hip_bf16.h>
#include <math.h>

// ---------------------------------------------------------------------------
// MultiheadAttention for MI455X (gfx1250, wave32, WMMA)
//   S=2048, B=2, D=512, E=512, H=8, hd=64
// Bias-bandwidth bound (134MB bias stream) -> flash-attention fused softmax,
// all GEMMs in f16 WMMA w/ f32 accumulate, B128 fragment loads, LDS-staged
// K/V tiles shared by 8 waves per workgroup.
// ---------------------------------------------------------------------------

#define SEQ      2048
#define BATCH    2
#define IN_DIM   512
#define EMBED    512
#define HEADS    8
#define HEAD_DIM 64
#define ROWS     (BATCH * SEQ)      // 4096
#define QKV_N    (3 * EMBED)        // 1536

typedef _Float16 v16h __attribute__((ext_vector_type(16)));
typedef _Float16 v8h  __attribute__((ext_vector_type(8)));
typedef float    v8f  __attribute__((ext_vector_type(8)));

// Per ISA 7.12.2, a 16-bit A/B fragment lane (half-group g = lane>>4) holds
// K = {g*8 .. g*8+7} in VGPRs 0-3 and K = {g*8+16 .. g*8+23} in VGPRs 4-7.
// Both runs are contiguous in K -> two 16-byte loads per fragment.
__device__ __forceinline__ v16h frag16(v8h lo, v8h hi) {
    return __builtin_shufflevector(lo, hi, 0, 1, 2, 3, 4, 5, 6, 7,
                                   8, 9, 10, 11, 12, 13, 14, 15);
}
__device__ __forceinline__ v16h load_frag_g(const _Float16* p) {   // global
    v8h lo = *(const v8h*)(p);
    v8h hi = *(const v8h*)(p + 16);
    return frag16(lo, hi);
}
__device__ __forceinline__ v16h load_frag_s(const _Float16* p) {   // LDS
    v8h lo = *(const v8h*)(p);
    v8h hi = *(const v8h*)(p + 16);
    return frag16(lo, hi);
}

// Row reductions across the 16 lanes sharing a C/D-layout row set.
__device__ __forceinline__ float rowmax16(float v) {
    v = fmaxf(v, __shfl_xor(v, 1, 32));
    v = fmaxf(v, __shfl_xor(v, 2, 32));
    v = fmaxf(v, __shfl_xor(v, 4, 32));
    v = fmaxf(v, __shfl_xor(v, 8, 32));
    return v;
}
__device__ __forceinline__ float rowsum16(float v) {
    v += __shfl_xor(v, 1, 32);
    v += __shfl_xor(v, 2, 32);
    v += __shfl_xor(v, 4, 32);
    v += __shfl_xor(v, 8, 32);
    return v;
}

// ---------------------------------------------------------------------------
// fp32 -> f16 conversions.  Weights are stored TRANSPOSED (n-major) so the
// B operand's K-run is contiguous -> B128 fragment loads.
// ---------------------------------------------------------------------------
__global__ void cvt_x_kernel(const float* __restrict__ x, _Float16* __restrict__ X16) {
    int idx = blockIdx.x * blockDim.x + threadIdx.x;   // over ROWS*IN_DIM
    int m = idx >> 9, k = idx & (IN_DIM - 1);
    int b = m >> 11, s = m & (SEQ - 1);                // m = b*SEQ + s
    X16[idx] = (_Float16)x[((size_t)s * BATCH + b) * IN_DIM + k];
}

// dstT[n*K + k] = src[k*N + n]
__global__ void cvt_wT_kernel(const float* __restrict__ src, _Float16* __restrict__ dstT,
                              int K, int N) {
    int idx = blockIdx.x * blockDim.x + threadIdx.x;   // over N*K, k fast
    int n = idx / K, k = idx % K;
    dstT[idx] = (_Float16)src[(size_t)k * N + n];
}

// ---------------------------------------------------------------------------
// QKV projection: [ROWS, 512] x [512, 1536] + bqkv
//   one wave -> one 16x16 output tile; K loop in steps of 32 (WMMA K).
//   Q scaled by 1/sqrt(2*hd); Q,K stored [B,H,S,hd]; V stored TRANSPOSED
//   [B,H,hd,S] so the PV B-operand K-run (keys) is contiguous.
// ---------------------------------------------------------------------------
__global__ void qkv_kernel(const _Float16* __restrict__ X16,
                           const _Float16* __restrict__ WT16,
                           const float* __restrict__ bqkv,
                           _Float16* __restrict__ Qf,
                           _Float16* __restrict__ Kf,
                           _Float16* __restrict__ Vt) {
    int wave = blockIdx.x * (blockDim.x >> 5) + (threadIdx.x >> 5);
    int lane = threadIdx.x & 31;
    int grp = lane >> 4, ln = lane & 15;
    int mt = wave / (QKV_N / 16);   // 0..255
    int nt = wave % (QKV_N / 16);   // 0..95

    const _Float16* arow = X16  + (size_t)(mt * 16 + ln) * IN_DIM + grp * 8;
    const _Float16* brow = WT16 + (size_t)(nt * 16 + ln) * IN_DIM + grp * 8;

    v8f acc = {};
    for (int k0 = 0; k0 < IN_DIM; k0 += 32) {
        v16h a  = load_frag_g(arow + k0);
        v16h bm = load_frag_g(brow + k0);
        acc = __builtin_amdgcn_wmma_f32_16x16x32_f16(false, a, false, bm,
                                                     (short)0, acc, false, false);
    }

    // n-tile is 16-wide and 16 | 64, so h / which / dbase are wave-uniform.
    int nbase = nt * 16;
    int h     = nbase / 192;
    int rem   = nbase % 192;
    int which = rem >> 6;             // 0=q 1=k 2=v
    int d     = (rem & 63) + ln;
    float bias = bqkv[nbase + ln];
    const float qscale = 0.08838834764831845f;   // 1/sqrt(2*HEAD_DIM)

#pragma unroll
    for (int r = 0; r < 8; ++r) {
        int m = mt * 16 + r + (grp << 3);
        int b = m >> 11, s = m & (SEQ - 1);
        int bh = b * HEADS + h;
        float v = acc[r] + bias;
        if (which == 0)
            Qf[((size_t)bh * SEQ + s) * HEAD_DIM + d] = (_Float16)(v * qscale);
        else if (which == 1)
            Kf[((size_t)bh * SEQ + s) * HEAD_DIM + d] = (_Float16)v;
        else
            Vt[((size_t)bh * HEAD_DIM + d) * SEQ + s] = (_Float16)v;
    }
}

// ---------------------------------------------------------------------------
// Fused attention.  One workgroup (8 waves) = one (b,h) and 128 q-rows;
// K/V tiles for 32 keys are cooperatively staged in LDS once per iteration
// and consumed by all 8 waves (8x less L2 traffic).  Bias streamed once.
// ---------------------------------------------------------------------------
__global__ void attn_kernel(const _Float16* __restrict__ Qf,
                            const _Float16* __restrict__ Kf,
                            const _Float16* __restrict__ Vt,
                            const float* __restrict__ bias,
                            _Float16* __restrict__ Of) {
    __shared__ __align__(16) _Float16 Ktile[32 * HEAD_DIM];      // [key][d]   4KB
    __shared__ __align__(16) _Float16 Vtile[HEAD_DIM * 32];      // [d][key]   4KB
    __shared__ __align__(16) _Float16 stage[8][16 * 32];         // P transpose 8KB

    int tid  = threadIdx.x;
    int w    = tid >> 5;
    int lane = tid & 31;
    int grp  = lane >> 4, ln = lane & 15;

    int bh     = blockIdx.x >> 4;          // 0..15 = b*HEADS + h
    int qblock = blockIdx.x & 15;          // 128 q-rows per block
    int b = bh >> 3, h = bh & 7;
    int qbase = qblock * 128 + w * 16;     // this wave's 16 q-rows

    const _Float16* Qbase  = Qf + (size_t)bh * SEQ * HEAD_DIM;
    const _Float16* Kbase  = Kf + (size_t)bh * SEQ * HEAD_DIM;
    const _Float16* Vtbase = Vt + (size_t)bh * HEAD_DIM * SEQ;
    const float*    Bbase  = bias + (size_t)h * SEQ * SEQ;

    // Q fragment (A operand), rows qbase + ln; two 32-wide K-chunks over hd=64
    const _Float16* qrow = Qbase + (size_t)(qbase + ln) * HEAD_DIM + grp * 8;
    v16h qa0 = load_frag_g(qrow);
    v16h qa1 = load_frag_g(qrow + 32);

    float mrow[8], lrow[8];
    v8f accO[4];
#pragma unroll
    for (int r = 0; r < 8; ++r) { mrow[r] = -1e30f; lrow[r] = 0.0f; }
#pragma unroll
    for (int t = 0; t < 4; ++t) accO[t] = (v8f){};

    // cooperative-copy indices (256 threads, 2048 halves per tile)
    int ck_key = tid >> 3, ck_d = (tid & 7) * 8;    // K tile: [key][d]
    int cv_d = tid >> 2, cv_k = (tid & 3) * 8;      // Vt tile: [d][key]

    for (int k0 = 0; k0 < SEQ; k0 += 32) {
        __syncthreads();   // previous iteration fully consumed
        *(v8h*)&Ktile[ck_key * HEAD_DIM + ck_d] =
            *(const v8h*)(Kbase + (size_t)(k0 + ck_key) * HEAD_DIM + ck_d);
        *(v8h*)&Vtile[cv_d * 32 + cv_k] =
            *(const v8h*)(Vtbase + (size_t)cv_d * SEQ + k0 + cv_k);
        __syncthreads();   // tiles ready

        // ---- S = Q K^T for 32 keys (two 16-key subtiles) ----
        const _Float16* kr0 = &Ktile[(ln)      * HEAD_DIM + grp * 8];
        const _Float16* kr1 = &Ktile[(16 + ln) * HEAD_DIM + grp * 8];
        v8f s0 = {}, s1 = {};
        s0 = __builtin_amdgcn_wmma_f32_16x16x32_f16(false, qa0, false, load_frag_s(kr0),
                                                    (short)0, s0, false, false);
        s0 = __builtin_amdgcn_wmma_f32_16x16x32_f16(false, qa1, false, load_frag_s(kr0 + 32),
                                                    (short)0, s0, false, false);
        s1 = __builtin_amdgcn_wmma_f32_16x16x32_f16(false, qa0, false, load_frag_s(kr1),
                                                    (short)0, s1, false, false);
        s1 = __builtin_amdgcn_wmma_f32_16x16x32_f16(false, qa1, false, load_frag_s(kr1 + 32),
                                                    (short)0, s1, false, false);

        // ---- bias add + joint online softmax over the 32 keys ----
#pragma unroll
        for (int r = 0; r < 8; ++r) {
            int q = qbase + r + (grp << 3);           // C/D layout: M = r + 8*grp
            const float* brow = Bbase + (size_t)q * SEQ + k0 + ln;
            float v0 = s0[r] + brow[0];
            float v1 = s1[r] + brow[16];
            float mx = rowmax16(fmaxf(v0, v1));
            float newm = fmaxf(mrow[r], mx);
            float alpha = __expf(mrow[r] - newm);
            float p0 = __expf(v0 - newm);
            float p1 = __expf(v1 - newm);
            lrow[r] = lrow[r] * alpha + rowsum16(p0 + p1);
            mrow[r] = newm;
#pragma unroll
            for (int t = 0; t < 4; ++t) accO[t][r] *= alpha;
            stage[w][(r + (grp << 3)) * 32 + ln]      = (_Float16)p0;
            stage[w][(r + (grp << 3)) * 32 + 16 + ln] = (_Float16)p1;
        }

        // ---- P (C layout -> A layout via per-wave LDS stage), O += P V ----
        v16h pa = load_frag_s(&stage[w][ln * 32 + grp * 8]);
#pragma unroll
        for (int t = 0; t < 4; ++t) {
            v16h vb = load_frag_s(&Vtile[(t * 16 + ln) * 32 + grp * 8]);
            accO[t] = __builtin_amdgcn_wmma_f32_16x16x32_f16(false, pa, false, vb,
                                                             (short)0, accO[t], false, false);
        }

        if (k0 + 32 < SEQ)
            __builtin_prefetch(Bbase + (size_t)(qbase + (grp << 3)) * SEQ + k0 + 32 + ln, 0, 1);
    }

    // ---- epilogue: O / l, scatter to [B, S, E] with E = h*64 + d ----
#pragma unroll
    for (int r = 0; r < 8; ++r) {
        int q = qbase + r + (grp << 3);
        float inv = 1.0f / lrow[r];
#pragma unroll
        for (int t = 0; t < 4; ++t) {
            float o = accO[t][r] * inv;
            Of[((size_t)(b * SEQ + q)) * EMBED + h * HEAD_DIM + t * 16 + ln] = (_Float16)o;
        }
    }
}

// ---------------------------------------------------------------------------
// Output projection: [ROWS, 512] x [512, 512]^T + bo -> fp32 d_out [B, S, E]
// ---------------------------------------------------------------------------
__global__ void oproj_kernel(const _Float16* __restrict__ O16,
                             const _Float16* __restrict__ WoT16,
                             const float* __restrict__ bo,
                             float* __restrict__ out) {
    int wave = blockIdx.x * (blockDim.x >> 5) + (threadIdx.x >> 5);
    int lane = threadIdx.x & 31;
    int grp = lane >> 4, ln = lane & 15;
    int mt = wave / (EMBED / 16);   // 0..255
    int nt = wave % (EMBED / 16);   // 0..31

    const _Float16* arow = O16   + (size_t)(mt * 16 + ln) * EMBED + grp * 8;
    const _Float16* brow = WoT16 + (size_t)(nt * 16 + ln) * EMBED + grp * 8;

    v8f acc = {};
    for (int k0 = 0; k0 < EMBED; k0 += 32) {
        v16h a  = load_frag_g(arow + k0);
        v16h bm = load_frag_g(brow + k0);
        acc = __builtin_amdgcn_wmma_f32_16x16x32_f16(false, a, false, bm,
                                                     (short)0, acc, false, false);
    }

    int n = nt * 16 + ln;
    float bias = bo[n];
#pragma unroll
    for (int r = 0; r < 8; ++r) {
        int m = mt * 16 + r + (grp << 3);
        out[(size_t)m * EMBED + n] = acc[r] + bias;
    }
}

// ---------------------------------------------------------------------------
// Host-side launch
// ---------------------------------------------------------------------------
extern "C" void kernel_launch(void* const* d_in, const int* in_sizes, int n_in,
                              void* d_out, int out_size, void* d_ws, size_t ws_size,
                              hipStream_t stream) {
    (void)in_sizes; (void)n_in; (void)out_size; (void)ws_size;

    const float* x    = (const float*)d_in[0];   // [S, B, D]
    const float* bias = (const float*)d_in[1];   // [1, H, S, S]
    const float* Wqkv = (const float*)d_in[2];   // [D, 3E]
    const float* bqkv = (const float*)d_in[3];   // [3E]
    const float* Wo   = (const float*)d_in[4];   // [E, E]
    const float* bo   = (const float*)d_in[5];   // [E]
    float* out = (float*)d_out;                  // [B, S, E]

    // Workspace layout (f16), ~22 MB total
    _Float16* X16   = (_Float16*)d_ws;
    _Float16* WqT16 = X16   + (size_t)ROWS * IN_DIM;
    _Float16* WoT16 = WqT16 + (size_t)IN_DIM * QKV_N;
    _Float16* Qf    = WoT16 + (size_t)EMBED * EMBED;
    _Float16* Kf    = Qf    + (size_t)BATCH * HEADS * SEQ * HEAD_DIM;
    _Float16* Vt    = Kf    + (size_t)BATCH * HEADS * SEQ * HEAD_DIM;
    _Float16* Of    = Vt    + (size_t)BATCH * HEADS * SEQ * HEAD_DIM;

    cvt_x_kernel <<<(ROWS * IN_DIM) / 256, 256, 0, stream>>>(x, X16);
    cvt_wT_kernel<<<(IN_DIM * QKV_N) / 256, 256, 0, stream>>>(Wqkv, WqT16, IN_DIM, QKV_N);
    cvt_wT_kernel<<<(EMBED * EMBED) / 256, 256, 0, stream>>>(Wo, WoT16, EMBED, EMBED);

    // 256 * 96 tiles, 8 waves/block
    qkv_kernel <<<(ROWS / 16) * (QKV_N / 16) / 8, 256, 0, stream>>>(X16, WqT16, bqkv, Qf, Kf, Vt);
    // 16 (b,h) * 16 q-blocks of 128 rows
    attn_kernel<<<BATCH * HEADS * (SEQ / 128), 256, 0, stream>>>(Qf, Kf, Vt, bias, Of);
    // 256 * 32 tiles, 8 waves/block
    oproj_kernel<<<(ROWS / 16) * (EMBED / 16) / 8, 256, 0, stream>>>(Of, WoT16, bo, out);
}
